// PhaseMultiHeadModel_14121852469303
// MI455X (gfx1250) — matrix operand order repaired and checked by reading, not compiled, not run on hardware
//
#include <hip/hip_runtime.h>
#include <hip/hip_bf16.h>
#include <math.h>

// ---------------------------------------------------------------------------
// PhaseMultiHeadModel on MI455X (gfx1250, wave32, WMMA).
// Final projection (2048x2048)@(2048x32000) = 268 GFLOP dominates; weights +
// logits ~0.5 GB -> ~23us of HBM at 23.3 TB/s, so everything runs through
// v_wmma_f32_16x16x32_f16 (f16 operands, f32 accum) fed from LDS tiles.
// CDNA5 async data movement: GLOBAL_LOAD_ASYNC_TO_LDS for the attention K/V
// tiles (ASYNCcnt) and a TDM tensor_load_to_lds descriptor for the GEMM A
// tile (TENSORcnt), per CDNA5 ISA.
// ---------------------------------------------------------------------------

typedef _Float16 f16;
typedef _Float16 v8h  __attribute__((ext_vector_type(8)));
typedef _Float16 v16h __attribute__((ext_vector_type(16)));
typedef float    v8f  __attribute__((ext_vector_type(8)));
typedef unsigned int v4u __attribute__((ext_vector_type(4)));
typedef unsigned int v8u __attribute__((ext_vector_type(8)));

static constexpr int kB  = 2;
static constexpr int kS  = 1024;
static constexpr int kV  = 32000;
static constexpr int kD  = 1024;
static constexpr int kH  = 16;
static constexpr int kHD = 64;

#define DEVFN static __device__ __forceinline__

DEVFN v16h cat16(v8h a, v8h b) {
  return __builtin_shufflevector(a, b, 0,1,2,3,4,5,6,7,8,9,10,11,12,13,14,15);
}
DEVFN v8f wmma_f16(v16h a, v16h b, v8f c) {
  // D = A(16x32 f16) * B(32x16 f16) + C(16x16 f32)
  return __builtin_amdgcn_wmma_f32_16x16x32_f16(
      /*neg_a=*/false, a, /*neg_b=*/false, b,
      /*c_mod=*/(short)0, c, /*reuse_a=*/false, /*reuse_b=*/false);
}
DEVFN v8f vzero8() {
  v8f z;
#pragma unroll
  for (int i = 0; i < 8; ++i) z[i] = 0.f;
  return z;
}

// --- CDNA5 async global->LDS copy (ASYNCcnt) with sync fallback -------------
#if __has_builtin(__builtin_amdgcn_global_load_async_to_lds_b128) && \
    __has_builtin(__builtin_amdgcn_s_wait_asynccnt)
#define USE_ASYNC_LDS 1
// Probe-learned signature: (addrspace(1) i32x4*, addrspace(3) i32x4*, imm, imm)
typedef int i32x4 __attribute__((vector_size(16)));
typedef __attribute__((address_space(1))) i32x4 as1_i32x4;
typedef __attribute__((address_space(3))) i32x4 as3_i32x4;
DEVFN void async_cp_b128(const void* g, void* l) {
  // LDS flat addr low 32 bits == LDS byte offset
  __builtin_amdgcn_global_load_async_to_lds_b128(
      (as1_i32x4*)(unsigned long long)(size_t)g,
      (as3_i32x4*)(size_t)(unsigned)(size_t)l, 0, 0);
}
DEVFN void wait_async0() { __builtin_amdgcn_s_wait_asynccnt(0); }
#else
#define USE_ASYNC_LDS 0
#endif

DEVFN void wait_tensor0() {
#if __has_builtin(__builtin_amdgcn_s_wait_tensorcnt)
  __builtin_amdgcn_s_wait_tensorcnt(0);
#else
  asm volatile("s_wait_tensorcnt 0x0" ::: "memory");
#endif
}

// --- TDM: issue a 2-D tile load (elem=2B) global -> LDS ---------------------
// D# per CDNA5 ISA: group0 {count, lds_addr, global
// addr, type=2}; group1 {data_size=1(2B), tensor dims, tile dims, stride};
// groups 2/3 zero (tile_dim3/4 unused). Issued once per wave; TENSORcnt.
DEVFN void tdm_load_tile_2d(const void* gsrc, void* ldst,
                            unsigned tile_x, unsigned tile_y,
                            unsigned row_stride_elems) {
  unsigned long long ga = (unsigned long long)(size_t)gsrc;
  unsigned lds = (unsigned)(size_t)ldst;
  v4u g0;
  g0[0] = 1u;                                            // count=1, user D#
  g0[1] = lds;                                           // lds_addr
  g0[2] = (unsigned)(ga & 0xffffffffu);                  // global_addr[31:0]
  g0[3] = (unsigned)((ga >> 32) & 0x01ffffffu) | (2u << 30); // [56:32] | type=2
  v8u g1;
  g1[0] = (1u << 16);                                    // data_size=2B, mask=0
  g1[1] = (tile_x & 0xffffu) << 16;                      // tensor_dim0[15:0]
  g1[2] = (tile_x >> 16) | ((tile_y & 0xffffu) << 16);   // td0[31:16]|td1[15:0]
  g1[3] = (tile_y >> 16) | (tile_x << 16);               // td1[31:16]|tile_dim0
  g1[4] = tile_y;                                        // tile_dim1 (dim2=0)
  g1[5] = row_stride_elems;                              // dim0_stride[31:0]
  g1[6] = 0u;                                            // stride hi | d1_stride
  g1[7] = 0u;
  v4u gz;
  gz[0] = 0u; gz[1] = 0u; gz[2] = 0u; gz[3] = 0u;
  asm volatile("tensor_load_to_lds %0, %1, %2, %3"
               :: "s"(g0), "s"(g1), "s"(gz), "s"(gz) : "memory");
}

// ---------------------------------------------------------------------------
// Stage 1: embedding -> tanh magnitude -> positional phase -> complex state,
// plus rotated q/k/v in f16. q,k layout: [b*H+h][s][128] ([re(64)|im(64)]).
// v layout TRANSPOSED per head: [b*H+h][dim(128)][t] so attention can stage
// the V tile for WMMA-B fragments with straight contiguous copies.
// ---------------------------------------------------------------------------
__global__ __launch_bounds__(256) void k_embed_phase(
    const int* __restrict__ x, const float* __restrict__ emb,
    const float* __restrict__ qrot, const float* __restrict__ krot,
    const float* __restrict__ vrot,
    float* __restrict__ sr, float* __restrict__ si,
    f16* __restrict__ qf, f16* __restrict__ kf, f16* __restrict__ vft)
{
  int idx = blockIdx.x * 256 + threadIdx.x;        // over B*S*D
  int d = idx & (kD - 1);
  int s = (idx >> 10) & (kS - 1);
  int b = idx >> 20;

  float e   = emb[(size_t)x[b * kS + s] * kD + d];
  float mag = tanhf(e);
  float freq  = __expf(-(float)d * (9.210340371976184f / (float)kD)); // ln(1e4)
  float phase = (float)s * freq * 3.14159265358979323f;

  float sp, cp;
  __sincosf(phase, &sp, &cp);
  sr[idx] = mag * cp;
  si[idx] = mag * sp;

  int h = d >> 6, hd = d & 63;
  int bh = b * kH + h;

  float sa, ca;
  __sincosf(phase + qrot[h * kHD + hd], &sa, &ca);
  size_t qb = ((size_t)bh * kS + s) * 128;
  qf[qb + hd]      = (f16)(mag * ca);
  qf[qb + 64 + hd] = (f16)(mag * sa);

  __sincosf(phase + krot[h * kHD + hd], &sa, &ca);
  kf[qb + hd]      = (f16)(mag * ca);
  kf[qb + 64 + hd] = (f16)(mag * sa);

  __sincosf(phase + vrot[h * kHD + hd], &sa, &ca);
  size_t vb0 = ((size_t)bh * 128 + hd) * kS + s;
  vft[vb0]           = (f16)(mag * ca);
  vft[vb0 + 64 * kS] = (f16)(mag * sa);
}

// ---------------------------------------------------------------------------
// Stage 2: causal flash attention, one wave32 per (b,h, 16-query tile).
// logits*8 with logit = Re(q.conj(k))/sqrt(64) == raw 128-dim dot product.
// Online softmax; P tile in f16 via LDS; P@V with WMMA. Residual add fused.
// K/V tiles staged with GLOBAL_LOAD_ASYNC_TO_LDS when available.
// ---------------------------------------------------------------------------
__global__ __launch_bounds__(32) void k_attention(
    const f16* __restrict__ qf, const f16* __restrict__ kf,
    const f16* __restrict__ vft,
    const float* __restrict__ sr, const float* __restrict__ si,
    float* __restrict__ ar, float* __restrict__ ai)
{
  __shared__ __align__(16) f16 kbuf[32][128];   // [key][dim]  == B[n][k] for Q@K^T
  __shared__ __align__(16) f16 vbuf[128][32];   // [dim][key]  == B[n][k] for P@V
  __shared__ __align__(16) f16 pbuf[16][32];    // [row][key]  == A for P@V

  const int lane  = threadIdx.x;
  const int qt    = blockIdx.x & 63;        // S/16 tiles
  const int bh    = blockIdx.x >> 6;        // 0..B*H-1
  const int b     = bh >> 4, h = bh & 15;
  const int qbase = qt * 16;
  const int hi    = lane >> 4;              // lane half
  const int m     = lane & 15;

  // Q A-fragments: 16x128 -> 4 chunks of K=32 (ISA 16-bit A 16x32 layout).
  v16h qa[4];
  const f16* qrow = qf + ((size_t)bh * kS + qbase + m) * 128;
#pragma unroll
  for (int c = 0; c < 4; ++c) {
    v8h lo = *(const v8h*)(qrow + 32 * c + 8 * hi);
    v8h hb = *(const v8h*)(qrow + 32 * c + 16 + 8 * hi);
    qa[c] = cat16(lo, hb);
  }

  v8f out[8];
#pragma unroll
  for (int g = 0; g < 8; ++g) out[g] = vzero8();
  float psum[8], rmax[8];
#pragma unroll
  for (int r = 0; r < 8; ++r) { psum[r] = 0.f; rmax[r] = -3.0e38f; }

  const int ntiles = qbase / 32 + 1;        // 32-key tiles needed (causal)
  for (int t = 0; t < ntiles; ++t) {
    const int kbase = t * 32;

    // Stage K tile (32x128) and V^T tile (128x32).
#pragma unroll
    for (int it = 0; it < 16; ++it) {
      int i8 = it * 32 + lane;
      int key = i8 >> 4, ch = i8 & 15;
      const f16* g = kf + ((size_t)bh * kS + kbase + key) * 128 + ch * 8;
#if USE_ASYNC_LDS
      async_cp_b128(g, &kbuf[key][ch * 8]);
#else
      *(v8h*)&kbuf[key][ch * 8] = *(const v8h*)g;
#endif
    }
#pragma unroll
    for (int it = 0; it < 16; ++it) {
      int i8 = it * 32 + lane;
      int dim = i8 >> 2, ch = i8 & 3;
      const f16* g = vft + ((size_t)bh * 128 + dim) * kS + kbase + ch * 8;
#if USE_ASYNC_LDS
      async_cp_b128(g, &vbuf[dim][ch * 8]);
#else
      *(v8h*)&vbuf[dim][ch * 8] = *(const v8h*)g;
#endif
    }
#if USE_ASYNC_LDS
    wait_async0();
#endif
    __syncthreads();

    // S = Q @ K^T for two 16-key sub-tiles (K-dim 128 = 4 chained WMMAs).
    v8f sacc[2];
#pragma unroll
    for (int j = 0; j < 2; ++j) {
      v8f s = vzero8();
#pragma unroll
      for (int c = 0; c < 4; ++c) {
        v8h lo = *(const v8h*)&kbuf[j * 16 + m][32 * c + 8 * hi];
        v8h hb = *(const v8h*)&kbuf[j * 16 + m][32 * c + 16 + 8 * hi];
        s = wmma_f16(qa[c], cat16(lo, hb), s);
      }
      sacc[j] = s;
    }

    // Online softmax: per-row max across both sub-tiles and 16 lanes.
    const int key0 = kbase + m, key1 = kbase + 16 + m;
#pragma unroll
    for (int r = 0; r < 8; ++r) {
      int qrowi = qbase + r + 8 * hi;
      float s0 = (key0 <= qrowi) ? sacc[0][r] : -3.0e38f;
      float s1 = (key1 <= qrowi) ? sacc[1][r] : -3.0e38f;
      float tm = fmaxf(s0, s1);
      tm = fmaxf(tm, __shfl_xor(tm, 1));
      tm = fmaxf(tm, __shfl_xor(tm, 2));
      tm = fmaxf(tm, __shfl_xor(tm, 4));
      tm = fmaxf(tm, __shfl_xor(tm, 8));
      float nm = fmaxf(rmax[r], tm);
      float sc = __expf(rmax[r] - nm);
      rmax[r] = nm;
      float p0 = (key0 <= qrowi) ? __expf(sacc[0][r] - nm) : 0.f;
      float p1 = (key1 <= qrowi) ? __expf(sacc[1][r] - nm) : 0.f;
      psum[r] = psum[r] * sc + p0 + p1;
#pragma unroll
      for (int g = 0; g < 8; ++g) out[g][r] *= sc;   // rescale accumulators
      pbuf[r + 8 * hi][m]      = (f16)p0;
      pbuf[r + 8 * hi][16 + m] = (f16)p1;
    }
    __syncthreads();

    // out += P(16x32) @ V(32x128): 8 WMMAs.
    {
      v8h plo = *(const v8h*)&pbuf[m][8 * hi];
      v8h phb = *(const v8h*)&pbuf[m][16 + 8 * hi];
      v16h pa = cat16(plo, phb);
#pragma unroll
      for (int g = 0; g < 8; ++g) {
        v8h lo = *(const v8h*)&vbuf[g * 16 + m][8 * hi];
        v8h hb = *(const v8h*)&vbuf[g * 16 + m][16 + 8 * hi];
        out[g] = wmma_f16(pa, cat16(lo, hb), out[g]);
      }
    }
    __syncthreads();
  }

  // Row sums across 16 lanes of each half, then normalize + residual + store.
#pragma unroll
  for (int r = 0; r < 8; ++r) {
    float v = psum[r];
    v += __shfl_xor(v, 1);
    v += __shfl_xor(v, 2);
    v += __shfl_xor(v, 4);
    v += __shfl_xor(v, 8);
    psum[r] = 1.f / v;
  }
#pragma unroll
  for (int g = 0; g < 8; ++g) {
    int col = g * 16 + m;                      // 0..127 within [re|im]
    int d = h * kHD + (col & 63);
    bool isImag = col >= 64;
#pragma unroll
    for (int r = 0; r < 8; ++r) {
      int srow = qbase + r + 8 * hi;
      size_t off = ((size_t)b * kS + srow) * kD + d;
      float val = out[g][r] * psum[r];
      if (isImag) ai[off] = si[off] + val;
      else        ar[off] = sr[off] + val;
    }
  }
}

// ---------------------------------------------------------------------------
// Generic WMMA GEMM: C(MxN) = [A1|A2](M x 2Kh) @ [B1; signB2*B2](2Kh x N)
//                              (+ bias1 + bias2).
// Block = 256 threads (8 waves), tile M=128, N=64, Kstep=32. A is
// pre-converted to a packed f16 mirror once (k_pack2) and then DMA'd per
// tile by the TDM (tensor_load_to_lds, TENSORcnt). B is converted+
// transposed+signed by the block (it differs per GEMM). Used for the complex
// FF and the 268-GFLOP output projection.
// ---------------------------------------------------------------------------
__global__ __launch_bounds__(256) void k_gemm(
    const f16* __restrict__ A12,   // packed f16 [M][2*Kh] ([A1|A2] pre-concat)
    const float* __restrict__ B1, const float* __restrict__ B2,
    float signB2,
    const float* __restrict__ bias1, const float* __restrict__ bias2,
    float* __restrict__ C, int M, int N, int Kh)
{
  __shared__ __align__(16) f16 Ab[128][32];
  __shared__ __align__(16) f16 Bt[64][32];    // transposed: [n][k]

  const int tid  = threadIdx.x;
  const int lane = tid & 31, w = tid >> 5;
  const int hi   = lane >> 4, m = lane & 15;
  const int row0 = blockIdx.y * 128;
  const int col0 = blockIdx.x * 64;
  const int K2   = 2 * Kh;

  v8f acc[4];
#pragma unroll
  for (int c = 0; c < 4; ++c) acc[c] = vzero8();

  const int ksteps = K2 >> 5;
  for (int ks = 0; ks < ksteps; ++ks) {
    int kg = ks * 32;
    bool second = kg >= Kh;
    const float* Bm = second ? B2 : B1;
    float sg = second ? signB2 : 1.f;
    int kk = second ? (kg - Kh) : kg;

    // A tile via TDM: 128 rows x 32 cols of f16, row stride K2 elems.
    if (w == 0) {
      tdm_load_tile_2d(A12 + (size_t)row0 * K2 + kg, &Ab[0][0],
                       32u, 128u, (unsigned)K2);
    }

    // Prefetch next B K-tile (global_prefetch_b8).
    if (ks + 1 < ksteps) {
      int kg2 = kg + 32;
      const float* Bn = (kg2 >= Kh) ? B2 : B1;
      int kk2 = (kg2 >= Kh) ? (kg2 - Kh) : kg2;
      __builtin_prefetch(Bn + (size_t)(kk2 + (tid >> 3)) * N + col0 + (tid & 7) * 8, 0, 0);
    }

    // Stage B (32x64) transposed to [n][k] with sign folded in.
#pragma unroll
    for (int i = 0; i < 8; ++i) {
      int idx = tid + 256 * i;
      int bn = idx & 63, bk = idx >> 6;
      Bt[bn][bk] = (f16)(sg * Bm[(size_t)(kk + bk) * N + col0 + bn]);
    }
    if (w == 0) wait_tensor0();
    __syncthreads();

    // Wave w: rows 16w..16w+15, 4 x (16x16) output fragments.
    v8h alo = *(const v8h*)&Ab[16 * w + m][8 * hi];
    v8h ahb = *(const v8h*)&Ab[16 * w + m][16 + 8 * hi];
    v16h a = cat16(alo, ahb);
#pragma unroll
    for (int c = 0; c < 4; ++c) {
      v8h blo = *(const v8h*)&Bt[16 * c + m][8 * hi];
      v8h bhb = *(const v8h*)&Bt[16 * c + m][16 + 8 * hi];
      acc[c] = wmma_f16(a, cat16(blo, bhb), acc[c]);
    }
    __syncthreads();
  }

#pragma unroll
  for (int c = 0; c < 4; ++c) {
    int col = col0 + 16 * c + m;
    float badd = 0.f;
    if (bias1) badd += bias1[col];
    if (bias2) badd += bias2[col];
#pragma unroll
    for (int r = 0; r < 8; ++r) {
      int row = row0 + 16 * w + r + 8 * hi;
      float v = acc[c][r] + badd;
      float* p = &C[(size_t)row * N + col];
      if (bias1) __builtin_nontemporal_store(v, p);  // 262MB logits: NT store
      else       *p = v;
    }
  }
}

// ---------------------------------------------------------------------------
// Stage 4: ComplexNorm (magnitude whitening with unbiased std, keep phase).
// One 256-thread block per (b,s) row of D=1024. In-place on the FF output.
// ---------------------------------------------------------------------------
__global__ __launch_bounds__(256) void k_cnorm(float* __restrict__ zr,
                                               float* __restrict__ zi)
{
  __shared__ float red[256];
  __shared__ float red2[256];
  const int tid = threadIdx.x;
  const size_t base = (size_t)blockIdx.x * kD;

  float mloc[4], s = 0.f, s2 = 0.f;
#pragma unroll
  for (int i = 0; i < 4; ++i) {
    int d = tid + 256 * i;
    float a = zr[base + d], b = zi[base + d];
    float mm = sqrtf(a * a + b * b);
    mloc[i] = mm; s += mm; s2 += mm * mm;
  }
  red[tid] = s; red2[tid] = s2;
  __syncthreads();
  for (int off = 128; off > 0; off >>= 1) {
    if (tid < off) { red[tid] += red[tid + off]; red2[tid] += red2[tid + off]; }
    __syncthreads();
  }
  float mean = red[0] * (1.f / (float)kD);
  float var  = (red2[0] - (float)kD * mean * mean) * (1.f / (float)(kD - 1));
  float stdv = sqrtf(fmaxf(var, 0.f));
#pragma unroll
  for (int i = 0; i < 4; ++i) {
    int d = tid + 256 * i;
    float mm = mloc[i];
    float sc = tanhf((mm - mean) / (stdv + 1e-5f)) / (mm + 1e-5f);
    zr[base + d] *= sc;
    zi[base + d] *= sc;
  }
}

// Pack [A1|A2] (two MxKh f32) into one f16 [M][2*Kh] buffer for TDM tiling.
__global__ __launch_bounds__(256) void k_pack2(const float* __restrict__ a1,
                                               const float* __restrict__ a2,
                                               f16* __restrict__ dst, int Kh)
{
  int i = blockIdx.x * 256 + threadIdx.x;      // over M*2Kh
  int row = i / (2 * Kh), c = i % (2 * Kh);
  float v = (c < Kh) ? a1[(size_t)row * Kh + c] : a2[(size_t)row * Kh + c - Kh];
  dst[i] = (f16)v;
}

// ---------------------------------------------------------------------------
extern "C" void kernel_launch(void* const* d_in, const int* in_sizes, int n_in,
                              void* d_out, int out_size, void* d_ws, size_t ws_size,
                              hipStream_t stream)
{
  (void)in_sizes; (void)n_in; (void)out_size; (void)ws_size;

  const int*   x    = (const int*)d_in[0];
  const float* emb  = (const float*)d_in[1];
  const float* qrot = (const float*)d_in[2];
  const float* krot = (const float*)d_in[3];
  const float* vrot = (const float*)d_in[4];
  const float* ffr  = (const float*)d_in[5];
  const float* ffi  = (const float*)d_in[6];
  const float* wr   = (const float*)d_in[7];
  const float* br   = (const float*)d_in[8];
  const float* wi   = (const float*)d_in[9];
  const float* bi   = (const float*)d_in[10];
  float* out = (float*)d_out;

  // Workspace layout (~84 MB total), 256B-aligned slices.
  char* ws = (char*)d_ws;
  size_t off = 0;
  auto alloc = [&](size_t bytes) -> void* {
    void* p = ws + off;
    off += (bytes + 255) & ~(size_t)255;
    return p;
  };
  const size_t BSD  = (size_t)kB * kS * kD;          // 2,097,152
  const size_t QKVH = (size_t)kB * kH * kS * 128;    // halves per q/k/v tensor
  float* sr  = (float*)alloc(BSD * 4);
  float* si  = (float*)alloc(BSD * 4);
  f16*   qf  = (f16*)alloc(QKVH * 2);
  f16*   kf  = (f16*)alloc(QKVH * 2);
  f16*   vft = (f16*)alloc(QKVH * 2);
  float* ar  = (float*)alloc(BSD * 4);               // state + attn_out (real)
  float* ai  = (float*)alloc(BSD * 4);
  float* fr  = (float*)alloc(BSD * 4);               // FF output (real)
  float* fi  = (float*)alloc(BSD * 4);
  f16*   ah  = (f16*)alloc(BSD * 2 * 2);             // packed [A1|A2] f16

  const int MS = kB * kS;                            // 2048 rows

  // 1) embed + phases + rotated q/k/v
  k_embed_phase<<<dim3((unsigned)(BSD / 256)), dim3(256), 0, stream>>>(
      x, emb, qrot, krot, vrot, sr, si, qf, kf, vft);

  // 2) causal flash attention + residual
  k_attention<<<dim3(kB * kH * (kS / 16)), dim3(32), 0, stream>>>(
      qf, kf, vft, sr, si, ar, ai);

  // 3) complex FF: out_r = Sr@Fr - Si@Fi ; out_i = Sr@Fi + Si@Fr
  k_pack2<<<dim3((unsigned)(BSD * 2 / 256)), dim3(256), 0, stream>>>(ar, ai, ah, kD);
  k_gemm<<<dim3(kD / 64, MS / 128), dim3(256), 0, stream>>>(
      ah, ffr, ffi, -1.f, nullptr, nullptr, fr, MS, kD, kD);
  k_gemm<<<dim3(kD / 64, MS / 128), dim3(256), 0, stream>>>(
      ah, ffi, ffr, +1.f, nullptr, nullptr, fi, MS, kD, kD);

  // 4) ComplexNorm (in place on fr/fi)
  k_cnorm<<<dim3(MS), dim3(256), 0, stream>>>(fr, fi);

  // 5) logits = Nr@Wr + Ni@Wi + br + bi  (dominant GEMM, 268 GFLOP)
  k_pack2<<<dim3((unsigned)(BSD * 2 / 256)), dim3(256), 0, stream>>>(fr, fi, ah, kD);
  k_gemm<<<dim3(kV / 64, MS / 128), dim3(256), 0, stream>>>(
      ah, wr, wi, +1.f, br, bi, out, MS, kV, kD);
}
